// GraphAttentionLayer_39204461478460
// MI455X (gfx1250) — compile-verified, matrix operand order checked by reading, not run on hardware
//
#include <hip/hip_runtime.h>
#include <math.h>
#include <stdint.h>

#define N_NODES 50000
#define N_EDGES 800000
#define IN_F    128
#define OUT_F   128
#define HEADS   4
#define HEAD_DIM 32

typedef __attribute__((ext_vector_type(16))) _Float16 v16h;
typedef __attribute__((ext_vector_type(8)))  float    v8f;
typedef __attribute__((ext_vector_type(4)))  int      v4i;

// --- optional gfx1250 async global->LDS path (guarded; falls back to ds stores)
#if defined(__has_builtin)
#if __has_builtin(__builtin_amdgcn_global_load_async_to_lds_b128) && \
    __has_builtin(__builtin_amdgcn_s_wait_asynccnt)
#define USE_ASYNC_LDS 1
#endif
#endif
#ifndef USE_ASYNC_LDS
#define USE_ASYNC_LDS 0
#endif

#if USE_ASYNC_LDS
typedef __attribute__((address_space(1))) v4i* gv4p;  // global v4i*
typedef __attribute__((address_space(3))) v4i* lv4p;  // LDS v4i*
__device__ inline void async_copy16(const float* g, float* l) {
  // generic->AS casts via integer (always legal; LDS addr = low 32 bits)
  __builtin_amdgcn_global_load_async_to_lds_b128(
      (gv4p)(uintptr_t)g, (lv4p)(uint32_t)(uintptr_t)l, 0, 0);
}
#endif

// ---------------------------------------------------------------------------
// Fragment loaders (ISA 7.12.2 16-bit layouts).
// A-frag / row-major B-frag: lane L -> row m=L%16 (or col n=L%16),
//   K runs: base=(L/16)*8 -> halves[0..7]=K kb..kb+7, halves[8..15]=kb+16..kb+23.
// Row-contiguous source: two 32-byte runs (float4 pairs -> cvt to f16).
// ---------------------------------------------------------------------------
__device__ inline v16h load_frag_rowmajor(const float* row, int kb) {
  const float4* p0 = (const float4*)(row + kb);
  const float4* p1 = (const float4*)(row + kb + 16);
  const float4 x0 = p0[0], x1 = p0[1], y0 = p1[0], y1 = p1[1];
  v16h a;
  a[0] = (_Float16)x0.x; a[1]  = (_Float16)x0.y; a[2]  = (_Float16)x0.z; a[3]  = (_Float16)x0.w;
  a[4] = (_Float16)x1.x; a[5]  = (_Float16)x1.y; a[6]  = (_Float16)x1.z; a[7]  = (_Float16)x1.w;
  a[8] = (_Float16)y0.x; a[9]  = (_Float16)y0.y; a[10] = (_Float16)y0.z; a[11] = (_Float16)y0.w;
  a[12] = (_Float16)y1.x; a[13] = (_Float16)y1.y; a[14] = (_Float16)y1.z; a[15] = (_Float16)y1.w;
  return a;
}

// B matrix stored [K][HEAD_DIM]: column `col` with stride HEAD_DIM.
__device__ inline v16h load_frag_strided(const float* col, int kb) {
  v16h b;
#pragma unroll
  for (int i = 0; i < 8; ++i) {
    b[i]     = (_Float16)col[(size_t)(kb + i) * HEAD_DIM];
    b[i + 8] = (_Float16)col[(size_t)(kb + 16 + i) * HEAD_DIM];
  }
  return b;
}

// ---------------------------------------------------------------------------
// Kernel 1: fused QKV projection.
// Block = 256 threads (8 waves) per 16-row node stripe. The 16x128 f32 x-tile
// is staged in LDS once (async-to-LDS when available), then each wave owns a
// fixed (head, col-half) and accumulates q/k/v 16x16 tiles: A-frag loaded
// once per k-step from LDS, 3 WMMAs per k-step -> 12 v_wmma per wave.
// ---------------------------------------------------------------------------
__global__ __launch_bounds__(256) void qkv_wmma(
    const float* __restrict__ x,
    const float* __restrict__ wq, const float* __restrict__ bq,
    const float* __restrict__ wk, const float* __restrict__ bk,
    const float* __restrict__ wv, const float* __restrict__ bv,
    float* __restrict__ qkv /* [3][H][N][32] */) {
  __shared__ float lds_x[16 * IN_F];  // 8 KB

  const int tid = threadIdx.x;
  const int m0  = blockIdx.x * 16;  // N divisible by 16
  const float* xtile = x + (size_t)m0 * IN_F;  // 16 contiguous rows

#if USE_ASYNC_LDS
  for (int c = tid; c < 16 * IN_F / 4; c += 256)
    async_copy16(xtile + 4 * c, lds_x + 4 * c);
  __builtin_amdgcn_s_wait_asynccnt(0);
#else
  for (int c = tid; c < 16 * IN_F / 4; c += 256)
    ((float4*)lds_x)[c] = ((const float4*)xtile)[c];
#endif
  __syncthreads();

  const int wave = tid >> 5;           // 0..7 -> (head, col-half)
  const int lane = tid & 31;
  const int m    = lane & 15;
  const int half = lane >> 4;
  const int n    = lane & 15;
  const int h    = wave >> 1;
  const int d0   = (wave & 1) * 16;

  const float* w0 = wq + (size_t)h * IN_F * HEAD_DIM + d0 + n;  // q column
  const float* w1 = wk + (size_t)h * IN_F * HEAD_DIM + d0 + n;  // k column
  const float* w2 = wv + (size_t)h * IN_F * HEAD_DIM + d0 + n;  // v column
  const float* arow = lds_x + (size_t)m * IN_F;

  v8f c0 = {}, c1 = {}, c2 = {};
#pragma unroll
  for (int kk = 0; kk < 4; ++kk) {
    const int kb = kk * 32 + half * 8;
    const v16h a = load_frag_rowmajor(arow, kb);
    const v16h b0 = load_frag_strided(w0, kb);
    c0 = __builtin_amdgcn_wmma_f32_16x16x32_f16(false, a, false, b0, (short)0, c0, false, false);
    const v16h b1 = load_frag_strided(w1, kb);
    c1 = __builtin_amdgcn_wmma_f32_16x16x32_f16(false, a, false, b1, (short)0, c1, false, false);
    const v16h b2 = load_frag_strided(w2, kb);
    c2 = __builtin_amdgcn_wmma_f32_16x16x32_f16(false, a, false, b2, (short)0, c2, false, false);
  }

  const float bq_ = bq[h * HEAD_DIM + d0 + n];
  const float bk_ = bk[h * HEAD_DIM + d0 + n];
  const float bv_ = bv[h * HEAD_DIM + d0 + n];
  float* dq = qkv + ((size_t)(0 * HEADS + h) * N_NODES) * HEAD_DIM;
  float* dk = qkv + ((size_t)(1 * HEADS + h) * N_NODES) * HEAD_DIM;
  float* dv = qkv + ((size_t)(2 * HEADS + h) * N_NODES) * HEAD_DIM;
#pragma unroll
  for (int r = 0; r < 8; ++r) {
    const size_t off = (size_t)(m0 + r + half * 8) * HEAD_DIM + d0 + n;
    dq[off] = c0[r] + bq_;
    dk[off] = c1[r] + bk_;
    dv[off] = c2[r] + bv_;
  }
}

// ---------------------------------------------------------------------------
// Kernel 2: init segment-max (-inf), denominators (0), aggregates (0).
// ---------------------------------------------------------------------------
__global__ void init_buffers(float* __restrict__ mx, float* __restrict__ den,
                             float* __restrict__ agg) {
  const size_t i = (size_t)blockIdx.x * blockDim.x + threadIdx.x;
  const size_t hn = (size_t)HEADS * N_NODES;
  if (i < hn) { mx[i] = -INFINITY; den[i] = 0.0f; }
  if (i < hn * HEAD_DIM) agg[i] = 0.0f;
}

// ---------------------------------------------------------------------------
// Kernel 3: per-edge, per-head scores: s = dot(q[tgt], k[src]).
// Node tables (25.6 MB) are L2-resident on the 192 MB L2.
// ---------------------------------------------------------------------------
__global__ void edge_scores(const float* __restrict__ q, const float* __restrict__ k,
                            const int* __restrict__ src, const int* __restrict__ tgt,
                            float* __restrict__ s) {
  const long idx = (long)blockIdx.x * blockDim.x + threadIdx.x;
  if (idx >= (long)N_EDGES * HEADS) return;
  const int h = (int)(idx & (HEADS - 1));
  const long e = idx >> 2;
  const int is = src[e];
  const int it = tgt[e];
  const float4* qr = (const float4*)(q + ((size_t)h * N_NODES + it) * HEAD_DIM);
  const float4* kr = (const float4*)(k + ((size_t)h * N_NODES + is) * HEAD_DIM);
  float acc = 0.0f;
#pragma unroll
  for (int i = 0; i < HEAD_DIM / 4; ++i) {
    const float4 a = qr[i], b = kr[i];
    acc += a.x * b.x + a.y * b.y + a.z * b.z + a.w * b.w;
  }
  s[(size_t)h * N_EDGES + e] = acc;
}

__device__ inline void atomicMaxFloat(float* addr, float val) {
  unsigned int* ua = (unsigned int*)addr;
  unsigned int old = *ua;
  while (true) {
    const float cur = __uint_as_float(old);
    if (cur >= val) break;
    const unsigned int assumed = old;
    old = atomicCAS(ua, assumed, __float_as_uint(val));
    if (old == assumed) break;
  }
}

// Kernel 4: segment max over edges sharing a target node.
__global__ void seg_max(const float* __restrict__ s, const int* __restrict__ tgt,
                        float* __restrict__ mx) {
  const long idx = (long)blockIdx.x * blockDim.x + threadIdx.x;
  if (idx >= (long)N_EDGES * HEADS) return;
  const int h = (int)(idx & (HEADS - 1));
  const long e = idx >> 2;
  atomicMaxFloat(&mx[(size_t)h * N_NODES + tgt[e]], s[(size_t)h * N_EDGES + e]);
}

// Kernel 5: exp(s - m[tgt]) in-place, accumulate denominators.
__global__ void exp_den(float* __restrict__ s, const int* __restrict__ tgt,
                        const float* __restrict__ mx, float* __restrict__ den) {
  const long idx = (long)blockIdx.x * blockDim.x + threadIdx.x;
  if (idx >= (long)N_EDGES * HEADS) return;
  const int h = (int)(idx & (HEADS - 1));
  const long e = idx >> 2;
  const int it = tgt[e];
  const float ex = __expf(s[(size_t)h * N_EDGES + e] - mx[(size_t)h * N_NODES + it]);
  s[(size_t)h * N_EDGES + e] = ex;
  atomicAdd(&den[(size_t)h * N_NODES + it], ex);
}

// Kernel 6: weighted aggregation agg[h][tgt] += (e/den[tgt]) * v[h][src].
__global__ void aggregate(const float* __restrict__ s, const int* __restrict__ src,
                          const int* __restrict__ tgt, const float* __restrict__ den,
                          const float* __restrict__ v, float* __restrict__ agg) {
  const long idx = (long)blockIdx.x * blockDim.x + threadIdx.x;
  if (idx >= (long)N_EDGES * HEADS) return;
  const int h = (int)(idx & (HEADS - 1));
  const long e = idx >> 2;
  const int is = src[e];
  const int it = tgt[e];
  const float alpha = s[(size_t)h * N_EDGES + e] / den[(size_t)h * N_NODES + it];
  const float4* vr = (const float4*)(v + ((size_t)h * N_NODES + is) * HEAD_DIM);
  float* ar = agg + ((size_t)h * N_NODES + it) * HEAD_DIM;
#pragma unroll
  for (int i = 0; i < HEAD_DIM / 4; ++i) {
    const float4 vv = vr[i];
    atomicAdd(&ar[i * 4 + 0], alpha * vv.x);
    atomicAdd(&ar[i * 4 + 1], alpha * vv.y);
    atomicAdd(&ar[i * 4 + 2], alpha * vv.z);
    atomicAdd(&ar[i * 4 + 3], alpha * vv.w);
  }
}

// ---------------------------------------------------------------------------
// Kernel 7: output projection out = concat(agg)[N,128] @ wo^T + bo.
// Block = 256 threads per 16-row stripe; the gathered concat-head tile
// (agg[(f>>5)*N+m][f&31], f=0..127) is staged into LDS once, then 8 waves
// each compute one 16x16 output column tile. B rows (wo) are contiguous.
// ---------------------------------------------------------------------------
__global__ __launch_bounds__(256) void out_wmma(
    const float* __restrict__ agg, const float* __restrict__ wo,
    const float* __restrict__ bo, float* __restrict__ out) {
  __shared__ float lds_a[16 * IN_F];  // 8 KB

  const int tid = threadIdx.x;
  const int m0  = blockIdx.x * 16;

  // gather concat-head layout: chunk c -> row m=c>>5, 4-float col chunk fc=c&31
  for (int c = tid; c < 16 * IN_F / 4; c += 256) {
    const int mm = c >> 5, fc = c & 31;
    const float* g = agg + ((size_t)(fc >> 3) * N_NODES + m0 + mm) * HEAD_DIM + (fc & 7) * 4;
    float* l = lds_a + (size_t)mm * IN_F + fc * 4;
#if USE_ASYNC_LDS
    async_copy16(g, l);
#else
    *(float4*)l = *(const float4*)g;
#endif
  }
#if USE_ASYNC_LDS
  __builtin_amdgcn_s_wait_asynccnt(0);
#endif
  __syncthreads();

  const int wave = tid >> 5;
  const int lane = tid & 31;
  const int m    = lane & 15;
  const int half = lane >> 4;
  const int n    = lane & 15;
  const int j0   = wave * 16;  // output column tile

  const float* arow = lds_a + (size_t)m * IN_F;
  const float* wrow = wo + (size_t)(j0 + n) * IN_F;  // x wo^T -> B(k,n)=wo[j0+n][k]

  v8f c = {};
#pragma unroll
  for (int kk = 0; kk < 4; ++kk) {
    const int kb = kk * 32 + half * 8;
    const v16h a = load_frag_rowmajor(arow, kb);
    const v16h b = load_frag_rowmajor(wrow, kb);
    c = __builtin_amdgcn_wmma_f32_16x16x32_f16(false, a, false, b, (short)0, c, false, false);
  }

  const float bb = bo[j0 + n];
#pragma unroll
  for (int r = 0; r < 8; ++r) {
    const int mm = r + half * 8;
    out[(size_t)(m0 + mm) * OUT_F + j0 + n] = c[r] + bb;
  }
}

extern "C" void kernel_launch(void* const* d_in, const int* in_sizes, int n_in,
                              void* d_out, int out_size, void* d_ws, size_t ws_size,
                              hipStream_t stream) {
  const float* x  = (const float*)d_in[0];
  const int*   ei = (const int*)d_in[1];
  const float* wq = (const float*)d_in[2];
  const float* bq = (const float*)d_in[3];
  const float* wk = (const float*)d_in[4];
  const float* bk = (const float*)d_in[5];
  const float* wv = (const float*)d_in[6];
  const float* bv = (const float*)d_in[7];
  const float* wo = (const float*)d_in[8];
  const float* bo = (const float*)d_in[9];
  float* out = (float*)d_out;

  // workspace carve-up (floats)
  float* ws  = (float*)d_ws;
  float* qkv = ws;                                            // 3*H*N*32
  float* sco = qkv + (size_t)3 * HEADS * N_NODES * HEAD_DIM;  // H*E
  float* mx  = sco + (size_t)HEADS * N_EDGES;                 // H*N
  float* den = mx  + (size_t)HEADS * N_NODES;                 // H*N
  float* agg = den + (size_t)HEADS * N_NODES;                 // H*N*32

  const float* q = qkv;
  const float* k = qkv + (size_t)HEADS * N_NODES * HEAD_DIM;
  const float* v = qkv + (size_t)2 * HEADS * N_NODES * HEAD_DIM;

  const int* src = ei;             // edge_index[0]
  const int* tgt = ei + N_EDGES;   // edge_index[1]

  // 1. QKV projection (WMMA, LDS-staged A): one block per 16-row stripe
  qkv_wmma<<<N_NODES / 16, 256, 0, stream>>>(x, wq, bq, wk, bk, wv, bv, qkv);

  // 2. init max/den/agg
  {
    const long tot = (long)HEADS * N_NODES * HEAD_DIM;
    init_buffers<<<(int)((tot + 255) / 256), 256, 0, stream>>>(mx, den, agg);
  }

  const long eh = (long)N_EDGES * HEADS;
  const int egrid = (int)((eh + 255) / 256);

  // 3..6: edge phases (L2-resident gathers + atomics)
  edge_scores<<<egrid, 256, 0, stream>>>(q, k, src, tgt, sco);
  seg_max<<<egrid, 256, 0, stream>>>(sco, tgt, mx);
  exp_den<<<egrid, 256, 0, stream>>>(sco, tgt, mx, den);
  aggregate<<<egrid, 256, 0, stream>>>(sco, src, tgt, den, v, agg);

  // 7. output projection (WMMA, LDS-staged A)
  out_wmma<<<N_NODES / 16, 256, 0, stream>>>(agg, wo, bo, out);
}
